// M3Surv_35167192220526
// MI455X (gfx1250) — matrix operand order repaired and checked by reading, not compiled, not run on hardware
//
#include <hip/hip_runtime.h>
#include <hip/hip_bf16.h>

typedef __attribute__((ext_vector_type(16))) __bf16       v16bf;
typedef __attribute__((ext_vector_type(8)))  float        v8f;
typedef __attribute__((ext_vector_type(4)))  int          v4i;
typedef __attribute__((ext_vector_type(8)))  int          v8i;
typedef __attribute__((ext_vector_type(4)))  unsigned int v4u;

#define DHID 256
#define GLOBAL_AS __attribute__((address_space(1)))
#define LDS_AS    __attribute__((address_space(3)))

// Optional gfx1250 async-load-to-LDS path (guarded so compile never breaks).
#if defined(__has_builtin)
#  if __has_builtin(__builtin_amdgcn_global_load_async_to_lds_b128)
#    define HAVE_ASYNC_LDS 1
#  endif
#  if __has_builtin(__builtin_amdgcn_tensor_load_to_lds)
#    define HAVE_TDM 1
#  endif
#endif
#ifndef HAVE_ASYNC_LDS
#  define HAVE_ASYNC_LDS 0
#endif
#ifndef HAVE_TDM
#  define HAVE_TDM 0
#endif
// TDM builtin arity differs across toolchains (probe doc): clang-22 = 5 args,
// clang-23+ = 6 args.
#if defined(__clang_major__) && (__clang_major__ >= 23)
#  define TDM_6ARG 1
#else
#  define TDM_6ARG 0
#endif

#if HAVE_ASYNC_LDS
__device__ __forceinline__ void async_copy16(const float* gsrc, float* lds_dst) {
  GLOBAL_AS v4i* gp = (GLOBAL_AS v4i*)(GLOBAL_AS void*)const_cast<float*>(gsrc);
  LDS_AS    v4i* lp = (LDS_AS    v4i*)(LDS_AS    void*)lds_dst;
  __builtin_amdgcn_global_load_async_to_lds_b128(gp, lp, 0, 0);
}
__device__ __forceinline__ void wait_async0() {
#  if defined(__has_builtin) && __has_builtin(__builtin_amdgcn_s_wait_asynccnt)
  __builtin_amdgcn_s_wait_asynccnt(0);
#  else
  asm volatile("s_wait_asynccnt 0" ::: "memory");
#  endif
}
#endif

// ---------------------------------------------------------------------------
// WMMA GEMM: C[M,N] = act(A[M,K] @ W[K,N] [+ bias] [+ residual])
// Block = 4 waves = 64x16 output supertile; W k-slab staged once in LDS
// (via gfx1250 async-to-LDS when available).  ACT: 0=none, 1=relu, 2=relu6.
// bf16 inputs, f32 accumulate.
// ---------------------------------------------------------------------------
template <int ACT, bool HASB, bool HASR>
__global__ __launch_bounds__(128) void gemm_wmma_kernel(
    const float* __restrict__ A, const float* __restrict__ W,
    const float* __restrict__ bias, const float* __restrict__ R,
    float* __restrict__ C, int M, int Kd, int Nn)
{
  __shared__ float Wsh[32 * 16];       // one k-slab of W: [k 0..31][n 0..15]

  const int tid  = threadIdx.x;        // 0..127
  const int wave = tid >> 5;           // 0..3 -> m-tile within supertile
  const int lane = tid & 31;
  const int half = lane >> 4;
  const int l15  = lane & 15;
  const int tn   = blockIdx.y;
  const int m0   = blockIdx.x * 64;

  int arow = m0 + wave * 16 + l15;
  if (arow >= M) arow = M - 1;         // clamp loads; stores masked
  const float* Arow = A + (size_t)arow * Kd;

  // staging assignment: thread -> (k row, 4-col group) of the 32x16 slab
  const int sk = tid >> 2;             // 0..31
  const int sn = (tid & 3) * 4;        // 0,4,8,12

  v8f acc = {};
  for (int kb = 0; kb < Kd; kb += 32) {
    __builtin_prefetch(Arow + kb + 64, 0, 1);     // global_prefetch
    // ---- stage W[kb..kb+31, tn*16..tn*16+15] into LDS (coalesced b128) ----
    {
      const float* src = W + (size_t)(kb + sk) * Nn + tn * 16 + sn;
#if HAVE_ASYNC_LDS
      async_copy16(src, &Wsh[sk * 16 + sn]);
#else
      *(float4*)&Wsh[sk * 16 + sn] = *(const float4*)src;
#endif
    }
#if HAVE_ASYNC_LDS
    wait_async0();
#endif
    __syncthreads();

    // ---- B fragment from LDS: column l15, k = half*16 + j ----
    v16bf b;
#pragma unroll
    for (int j = 0; j < 16; ++j)
      b[j] = (__bf16)Wsh[(half * 16 + j) * 16 + l15];

    // ---- A fragment: lane holds row l15; two runs of 8 k-values ----
    v16bf a;
    const int ka0 = kb + half * 16;
    const int ka1 = kb + half * 16 + 8;
#pragma unroll
    for (int j = 0; j < 8; ++j) a[j]     = (__bf16)Arow[ka0 + j];
#pragma unroll
    for (int j = 0; j < 8; ++j) a[8 + j] = (__bf16)Arow[ka1 + j];

    acc = __builtin_amdgcn_wmma_f32_16x16x32_bf16(false, a, false, b,
                                                  (short)0, acc, false, false);
    __syncthreads();                   // slab reuse barrier
  }

  const int col = tn * 16 + l15;
  const float bv = HASB ? bias[col] : 0.0f;
#pragma unroll
  for (int r = 0; r < 8; ++r) {
    const int row = m0 + wave * 16 + r + half * 8;
    if (row < M) {
      float v = acc[r] + bv;
      if (HASR) v += R[(size_t)row * Nn + col];
      if (ACT == 1)      v = fmaxf(v, 0.0f);
      else if (ACT == 2) v = fminf(fmaxf(v, 0.0f), 6.0f);
      C[(size_t)row * Nn + col] = v;
    }
  }
}

// ---------------------------------------------------------------------------
// Flash-style attention: one wave per (16-query tile, head). DH = 32.
// Q/K/V are [n, 256] with head h at columns h*32 .. h*32+31.
// Online softmax; LDS used to transpose scores (C layout) into P (A layout).
// ---------------------------------------------------------------------------
__global__ __launch_bounds__(32) void flash_attn_kernel(
    const float* __restrict__ Q, const float* __restrict__ Km,
    const float* __restrict__ V, float* __restrict__ O, int nq, int nk)
{
  __shared__ float Ssh[16 * 32];
  __shared__ float rscale[16];
  __shared__ float rl[16];

  const int lane = threadIdx.x;
  const int half = lane >> 4;
  const int l15  = lane & 15;
  const int tm = blockIdx.x;
  const int h  = blockIdx.y;

  int qrow = tm * 16 + l15;
  if (qrow >= nq) qrow = nq - 1;
  const float* Qrow = Q + (size_t)qrow * DHID + h * 32;

  v16bf qa;
#pragma unroll
  for (int j = 0; j < 8; ++j) qa[j]     = (__bf16)Qrow[half * 16 + j];
#pragma unroll
  for (int j = 0; j < 8; ++j) qa[8 + j] = (__bf16)Qrow[half * 16 + 8 + j];

  float mrow = -1e30f, lrow = 0.0f;   // stats for query row l15 (dup per half)
  v8f o0 = {}, o1 = {};
  const float sc = 0.17677669529663687f;  // 1/sqrt(32)

  for (int kb = 0; kb < nk; kb += 32) {
    // ---- scores for 32 keys: two WMMAs (Q 16x32) x (K^T 32x16)
    int k0i = kb + l15;        if (k0i >= nk) k0i = nk - 1;
    int k1i = kb + 16 + l15;   if (k1i >= nk) k1i = nk - 1;
    const float* K0 = Km + (size_t)k0i * DHID + h * 32 + half * 16;
    const float* K1 = Km + (size_t)k1i * DHID + h * 32 + half * 16;
    v16bf b0, b1;
#pragma unroll
    for (int j = 0; j < 16; ++j) { b0[j] = (__bf16)K0[j]; b1[j] = (__bf16)K1[j]; }
    v8f s0 = {}, s1 = {};
    s0 = __builtin_amdgcn_wmma_f32_16x16x32_bf16(false, qa, false, b0, (short)0, s0, false, false);
    s1 = __builtin_amdgcn_wmma_f32_16x16x32_bf16(false, qa, false, b1, (short)0, s1, false, false);

    // ---- dump scaled + masked scores to LDS (C layout -> row-major)
#pragma unroll
    for (int r = 0; r < 8; ++r) {
      const int row = r + half * 8;
      Ssh[row * 32 + l15]      = (kb + l15      < nk) ? s0[r] * sc : -1e30f;
      Ssh[row * 32 + 16 + l15] = (kb + 16 + l15 < nk) ? s1[r] * sc : -1e30f;
    }
    __syncthreads();

    // ---- online softmax for row l15 (duplicated by both halves)
    float tmax = -1e30f;
    for (int j = 0; j < 32; ++j) tmax = fmaxf(tmax, Ssh[l15 * 32 + j]);
    const float mnew = fmaxf(mrow, tmax);
    const float scl  = __expf(mrow - mnew);
    float psum = 0.0f;
    for (int j = 0; j < 32; ++j) {
      const float pexp = __expf(Ssh[l15 * 32 + j] - mnew);
      psum += pexp;
      Ssh[l15 * 32 + j] = pexp;       // overwrite scores with probabilities
    }
    lrow = lrow * scl + psum;
    mrow = mnew;
    rscale[l15] = scl;
    __syncthreads();

    // ---- rescale output accumulators
#pragma unroll
    for (int r = 0; r < 8; ++r) {
      const float f = rscale[r + half * 8];
      o0[r] *= f; o1[r] *= f;
    }

    // ---- P (A layout from LDS) x V (B layout from global): two WMMAs
    v16bf pa;
#pragma unroll
    for (int j = 0; j < 8; ++j) pa[j]     = (__bf16)Ssh[l15 * 32 + half * 16 + j];
#pragma unroll
    for (int j = 0; j < 8; ++j) pa[8 + j] = (__bf16)Ssh[l15 * 32 + half * 16 + 8 + j];

    v16bf v0, v1;
#pragma unroll
    for (int j = 0; j < 16; ++j) {
      int key = kb + half * 16 + j;
      if (key >= nk) key = nk - 1;    // p==0 there anyway
      const float* Vr = V + (size_t)key * DHID + h * 32;
      v0[j] = (__bf16)Vr[l15];
      v1[j] = (__bf16)Vr[16 + l15];
    }
    o0 = __builtin_amdgcn_wmma_f32_16x16x32_bf16(false, pa, false, v0, (short)0, o0, false, false);
    o1 = __builtin_amdgcn_wmma_f32_16x16x32_bf16(false, pa, false, v1, (short)0, o1, false, false);
    __syncthreads();
  }

  rl[l15] = lrow;
  __syncthreads();
#pragma unroll
  for (int r = 0; r < 8; ++r) {
    const int row = tm * 16 + r + half * 8;
    if (row < nq) {
      const float inv = 1.0f / rl[r + half * 8];
      O[(size_t)row * DHID + h * 32 + l15]      = o0[r] * inv;
      O[(size_t)row * DHID + h * 32 + 16 + l15] = o1[r] * inv;
    }
  }
}

// ---------------------------------------------------------------------------
// Hypergraph edge gather-mean (fixed arity K=16).
// TDM path: one gather-mode tensor_load_to_lds pulls the 16 node rows
// (16-bit row indices, row width 256 f32) into LDS; block reduces the tile.
// Fallback: direct VMEM gather.
// ---------------------------------------------------------------------------
__global__ __launch_bounds__(256) void edge_mean_kernel(
    const float* __restrict__ y, const int* __restrict__ nodes,
    float* __restrict__ ef, int nrows)
{
#if HAVE_TDM
  __shared__ float Lsh[16 * DHID];     // gathered 16 rows x 256 cols (16 KB)
  const int e = blockIdx.x;
  const int c = threadIdx.x;
  if (threadIdx.x < 32) {              // wave 0 issues the TDM op
    const int* nd = nodes + e * 16;
    int idx[8];
#pragma unroll
    for (int j = 0; j < 8; ++j) {
      unsigned lo = (unsigned)nd[2 * j]     & 0xFFFFu;
      unsigned hi = (unsigned)nd[2 * j + 1] & 0xFFFFu;
      idx[j] = (int)(lo | (hi << 16));
    }
    unsigned lds_off = (unsigned)(unsigned long long)(LDS_AS void*)&Lsh[0];
    unsigned long long ga = (unsigned long long)y;
    v4u g0;
    g0[0] = 1u | (1u << 31);           // count=1, 16-bit indices, gather_mode=1
    g0[1] = lds_off;                   // lds_addr
    g0[2] = (unsigned)(ga & 0xFFFFFFFFu);                 // global_addr[31:0]
    g0[3] = (unsigned)((ga >> 32) & 0x1FFFFFFu) | (2u << 30);  // addr hi, type=2
    v8i g1 = {};
    g1[0] = (2 << 16);                 // data_size = 4 bytes
    g1[1] = (int)(((unsigned)DHID & 0xFFFFu) << 16);      // tensor_dim0 lo16
    g1[2] = (int)(((unsigned)nrows & 0xFFFFu) << 16);     // dim0 hi=0 | dim1 lo16
    g1[3] = (int)(((unsigned)DHID & 0xFFFFu) << 16);      // dim1 hi=0 | tile_dim0
    g1[4] = 16;                        // tile_dim1 = #valid indices
    g1[5] = DHID;                      // tensor_dim0_stride lo32
    g1[6] = 0;
    g1[7] = 0;
    v4i g2; g2[0] = idx[0]; g2[1] = idx[1]; g2[2] = idx[2]; g2[3] = idx[3];
    v4i g3; g3[0] = idx[4]; g3[1] = idx[5]; g3[2] = idx[6]; g3[3] = idx[7];
#  if TDM_6ARG
    v8i g4 = {};
    __builtin_amdgcn_tensor_load_to_lds(g0, g1, g2, g3, g4, 0);
#  else
    __builtin_amdgcn_tensor_load_to_lds(g0, g1, g2, g3, 0);
#  endif
#  if defined(__has_builtin) && __has_builtin(__builtin_amdgcn_s_wait_tensorcnt)
    __builtin_amdgcn_s_wait_tensorcnt(0);
#  else
    asm volatile("s_wait_tensorcnt 0" ::: "memory");
#  endif
  }
  __syncthreads();
  float s = 0.0f;
#pragma unroll
  for (int r = 0; r < 16; ++r) s += Lsh[r * DHID + c];
  ef[(size_t)e * DHID + c] = s * (1.0f / 16.0f);
#else
  (void)nrows;
  const int e = blockIdx.x;
  const int c = threadIdx.x;
  const int* nd = nodes + e * 16;
  float s = 0.0f;
#pragma unroll
  for (int j = 0; j < 16; ++j) s += y[(size_t)nd[j] * DHID + c];
  ef[(size_t)e * DHID + c] = s * (1.0f / 16.0f);
#endif
}

__global__ __launch_bounds__(256) void scatter_kernel(
    const float* __restrict__ ef, const int* __restrict__ nodes,
    float* __restrict__ contrib, float* __restrict__ deg)
{
  const int m = blockIdx.x;
  const int c = threadIdx.x;
  const int v = nodes[m];
  atomicAdd(&contrib[(size_t)v * DHID + c], ef[(size_t)(m >> 4) * DHID + c]);
  if (c == 0) atomicAdd(&deg[v], 1.0f);
}

__global__ __launch_bounds__(256) void hg_finalize_kernel(
    const float* __restrict__ contrib, const float* __restrict__ deg,
    float* __restrict__ out)
{
  const int v = blockIdx.x;
  const int c = threadIdx.x;
  out[(size_t)v * DHID + c] = contrib[(size_t)v * DHID + c] / fmaxf(deg[v], 1.0f);
}

// ---------------------------------------------------------------------------
// Per-pathway SNN towers: gen[p] = elu(elu(x[p]@w1[p]+b1[p]) @ w2[p] + b2[p])
// ---------------------------------------------------------------------------
__global__ __launch_bounds__(256) void transomic_kernel(
    const float* __restrict__ x, const float* __restrict__ w1,
    const float* __restrict__ b1, const float* __restrict__ w2,
    const float* __restrict__ b2, float* __restrict__ gen)
{
  __shared__ float hsh[256];
  const int p = blockIdx.x;
  const int d = threadIdx.x;
  const float* xp  = x + (size_t)p * 2048;
  const float* w1p = w1 + (size_t)p * 2048 * DHID;
  float s = 0.0f;
  for (int i = 0; i < 2048; ++i) s += xp[i] * w1p[(size_t)i * DHID + d];
  s += b1[p * DHID + d];
  s = (s > 0.0f) ? s : (__expf(s) - 1.0f);
  hsh[d] = s;
  __syncthreads();
  const float* w2p = w2 + (size_t)p * DHID * DHID;
  float t = 0.0f;
  for (int i = 0; i < DHID; ++i) t += hsh[i] * w2p[i * DHID + d];
  t += b2[p * DHID + d];
  t = (t > 0.0f) ? t : (__expf(t) - 1.0f);
  gen[p * DHID + d] = t;
}

// ---------------------------------------------------------------------------
// LayerNorm over D=256 per row (in place)
// ---------------------------------------------------------------------------
__global__ __launch_bounds__(256) void layernorm_kernel(
    float* __restrict__ tok, const float* __restrict__ g,
    const float* __restrict__ b)
{
  __shared__ float red[256];
  const int r = blockIdx.x;
  const int c = threadIdx.x;
  const float x = tok[(size_t)r * DHID + c];
  red[c] = x;
  __syncthreads();
  for (int s = 128; s > 0; s >>= 1) {
    if (c < s) red[c] += red[c + s];
    __syncthreads();
  }
  const float mu = red[0] * (1.0f / 256.0f);
  __syncthreads();
  const float d = x - mu;
  red[c] = d * d;
  __syncthreads();
  for (int s = 128; s > 0; s >>= 1) {
    if (c < s) red[c] += red[c + s];
    __syncthreads();
  }
  const float var = red[0] * (1.0f / 256.0f);
  tok[(size_t)r * DHID + c] = d * rsqrtf(var + 1e-5f) * g[c] + b[c];
}

__global__ __launch_bounds__(256) void mean_embed_kernel(
    const float* __restrict__ tok, float* __restrict__ ge, float* __restrict__ pe)
{
  const int c = threadIdx.x;
  float s = 0.0f;
  for (int r = 0; r < 6; ++r) s += tok[r * DHID + c];
  ge[c] = s * (1.0f / 6.0f);
  float t = 0.0f;
  for (int r = 6; r < 4102; ++r) t += tok[(size_t)r * DHID + c];
  pe[c] = t * (1.0f / 4096.0f);
}

__global__ __launch_bounds__(128) void head_kernel(
    const float* __restrict__ ge, const float* __restrict__ pe,
    const float* __restrict__ mm_w, const float* __restrict__ mm_b,
    const float* __restrict__ cls_w, const float* __restrict__ cls_b,
    float* __restrict__ out)
{
  __shared__ float fus[128];
  const int j = threadIdx.x;
  float s = 0.0f;
  for (int i = 0; i < 256; ++i) s += ge[i] * mm_w[i * 128 + j];
  for (int i = 0; i < 256; ++i) s += pe[i] * mm_w[(256 + i) * 128 + j];
  s += mm_b[j];
  fus[j] = fminf(fmaxf(s, 0.0f), 6.0f);
  __syncthreads();
  if (j < 4) {
    float t = 0.0f;
    for (int i = 0; i < 128; ++i) t += fus[i] * cls_w[i * 4 + j];
    out[j] = t + cls_b[j];
  }
}

// ---------------------------------------------------------------------------
extern "C" void kernel_launch(void* const* d_in, const int* in_sizes, int n_in,
                              void* d_out, int out_size, void* d_ws, size_t ws_size,
                              hipStream_t stream) {
  (void)in_sizes; (void)n_in; (void)out_size; (void)ws_size;
  const float* x_ff      = (const float*)d_in[0];
  const float* x_ffpe    = (const float*)d_in[1];
  const float* x_trans   = (const float*)d_in[2];
  const float* ff_w1     = (const float*)d_in[3];
  const float* ff_b1     = (const float*)d_in[4];
  const float* ff_w2     = (const float*)d_in[5];
  const float* ff_b2     = (const float*)d_in[6];
  const float* fp_w1     = (const float*)d_in[7];
  const float* fp_b1     = (const float*)d_in[8];
  const float* fp_w2     = (const float*)d_in[9];
  const float* fp_b2     = (const float*)d_in[10];
  const float* trans_w1  = (const float*)d_in[11];
  const float* trans_b1  = (const float*)d_in[12];
  const float* trans_w2  = (const float*)d_in[13];
  const float* trans_b2  = (const float*)d_in[14];
  const float* gff_w     = (const float*)d_in[15];
  const float* gff_b     = (const float*)d_in[16];
  const float* gfp_w     = (const float*)d_in[17];
  const float* gfp_b     = (const float*)d_in[18];
  const float* gsh_w     = (const float*)d_in[19];
  const float* gsh_b     = (const float*)d_in[20];
  const float* sg_wq     = (const float*)d_in[21];
  const float* sg_wk     = (const float*)d_in[22];
  const float* sg_wv     = (const float*)d_in[23];
  const float* sg_wo     = (const float*)d_in[24];
  const float* af_wq     = (const float*)d_in[25];
  const float* af_wk     = (const float*)d_in[26];
  const float* af_wv     = (const float*)d_in[27];
  const float* af_wo     = (const float*)d_in[28];
  const float* ffn_w1    = (const float*)d_in[29];
  const float* ffn_b1    = (const float*)d_in[30];
  const float* ffn_w2    = (const float*)d_in[31];
  const float* ffn_b2    = (const float*)d_in[32];
  const float* ln_g      = (const float*)d_in[33];
  const float* ln_b      = (const float*)d_in[34];
  const float* mm_w      = (const float*)d_in[35];
  const float* mm_b      = (const float*)d_in[36];
  const float* cls_w     = (const float*)d_in[37];
  const float* cls_b     = (const float*)d_in[38];
  const int*   ff_nodes  = (const int*)d_in[39];
  const int*   fp_nodes  = (const int*)d_in[40];
  const int*   sh_nodes  = (const int*)d_in[41];

  const int N = 2048, D = 256, MTOT = 65536, E = 4096;

  // workspace slots (4224 rows x 256 cols each)
  const size_t SLOT = (size_t)4224 * 256;
  float* ws  = (float*)d_ws;
  float* ff  = ws + 0 * SLOT;
  float* fp  = ws + 1 * SLOT;
  float* pbf = ws + 2 * SLOT;   // shared-graph features p [4096,256]
  float* tA  = ws + 3 * SLOT;   // y / Q
  float* tB  = ws + 4 * SLOT;   // ef / K
  float* tC  = ws + 5 * SLOT;   // contrib / V
  float* sml = ws + 6 * SLOT;   // small area
  float* deg = sml;             // 4096
  float* gen = sml + 8192;      // 6*256
  float* ge  = sml + 16384;     // 256
  float* pe  = sml + 16640;     // 256
  float* aO  = ws + 7 * SLOT;   // attention output
  float* path= ws + 8 * SLOT;   // post-ShareGuide concat [4096,256]
  float* tok = ws + 9 * SLOT;   // [4102,256]
  float* hid = ws + 10 * SLOT;  // FFN hidden [4102,1024] (spans slots 10..13)

  auto gemm = [&](const float* A, const float* W, const float* bias,
                  const float* R, float* C, int M, int Kd, int Nn, int act) {
    dim3 grid((M + 63) / 64, Nn / 16);
    if (act == 2)
      gemm_wmma_kernel<2, true, false><<<grid, 128, 0, stream>>>(A, W, bias, R, C, M, Kd, Nn);
    else if (act == 1)
      gemm_wmma_kernel<1, true, false><<<grid, 128, 0, stream>>>(A, W, bias, R, C, M, Kd, Nn);
    else if (R && bias)
      gemm_wmma_kernel<0, true, true><<<grid, 128, 0, stream>>>(A, W, bias, R, C, M, Kd, Nn);
    else if (R)
      gemm_wmma_kernel<0, false, true><<<grid, 128, 0, stream>>>(A, W, bias, R, C, M, Kd, Nn);
    else
      gemm_wmma_kernel<0, false, false><<<grid, 128, 0, stream>>>(A, W, bias, R, C, M, Kd, Nn);
  };

  auto hgnnp = [&](float* x, const float* w, const float* b, const int* nodes, int n) {
    gemm(x, w, b, nullptr, tA, n, D, D, /*relu*/1);
    edge_mean_kernel<<<E, 256, 0, stream>>>(tA, nodes, tB, n);
    (void)hipMemsetAsync(tC, 0, (size_t)n * D * sizeof(float), stream);
    (void)hipMemsetAsync(deg, 0, (size_t)n * sizeof(float), stream);
    scatter_kernel<<<MTOT, 256, 0, stream>>>(tB, nodes, tC, deg);
    hg_finalize_kernel<<<n, 256, 0, stream>>>(tC, deg, x);
  };

  // ---- patch towers (Linear -> ReLU6 twice) ----
  gemm(x_ff,   ff_w1, ff_b1, nullptr, tA, N, 1024, D, 2);
  gemm(tA,     ff_w2, ff_b2, nullptr, ff, N, D,    D, 2);
  gemm(x_ffpe, fp_w1, fp_b1, nullptr, tA, N, 1024, D, 2);
  gemm(tA,     fp_w2, fp_b2, nullptr, fp, N, D,    D, 2);
  (void)hipMemcpyAsync(pbf,                 ff, (size_t)N * D * sizeof(float),
                       hipMemcpyDeviceToDevice, stream);
  (void)hipMemcpyAsync(pbf + (size_t)N * D, fp, (size_t)N * D * sizeof(float),
                       hipMemcpyDeviceToDevice, stream);

  // ---- 3 stacked hypergraph conv layers per branch ----
  for (int l = 0; l < 3; ++l) {
    hgnnp(ff,  gff_w + (size_t)l * D * D, gff_b + l * D, ff_nodes, N);
    hgnnp(fp,  gfp_w + (size_t)l * D * D, gfp_b + l * D, fp_nodes, N);
    hgnnp(pbf, gsh_w + (size_t)l * D * D, gsh_b + l * D, sh_nodes, 2 * N);
  }

  // ---- ShareGuide cross-attention (shared K/V from p) ----
  gemm(pbf, sg_wk, nullptr, nullptr, tB, 2 * N, D, D, 0);
  gemm(pbf, sg_wv, nullptr, nullptr, tC, 2 * N, D, D, 0);
  gemm(ff,  sg_wq, nullptr, nullptr, tA, N, D, D, 0);
  flash_attn_kernel<<<dim3(N / 16, 8), 32, 0, stream>>>(tA, tB, tC, aO, N, 2 * N);
  gemm(aO, sg_wo, nullptr, ff, path, N, D, D, 0);
  gemm(fp,  sg_wq, nullptr, nullptr, tA, N, D, D, 0);
  flash_attn_kernel<<<dim3(N / 16, 8), 32, 0, stream>>>(tA, tB, tC, aO, N, 2 * N);
  gemm(aO, sg_wo, nullptr, fp, path + (size_t)N * D, N, D, D, 0);

  // ---- transomic SNN towers ----
  transomic_kernel<<<6, 256, 0, stream>>>(x_trans, trans_w1, trans_b1,
                                          trans_w2, trans_b2, gen);

  // ---- AlignFusion: gene_c = gen + mha(gen, path, path) ----
  gemm(path, af_wk, nullptr, nullptr, tB, 2 * N, D, D, 0);
  gemm(path, af_wv, nullptr, nullptr, tC, 2 * N, D, D, 0);
  gemm(gen,  af_wq, nullptr, nullptr, tA, 6, D, D, 0);
  flash_attn_kernel<<<dim3(1, 8), 32, 0, stream>>>(tA, tB, tC, aO, 6, 2 * N);
  gemm(aO, af_wo, nullptr, gen, tok, 6, D, D, 0);

  // ---- AlignFusion: path_c = path + mha(path, gen, gen) ----
  gemm(gen,  af_wk, nullptr, nullptr, tB, 6, D, D, 0);
  gemm(gen,  af_wv, nullptr, nullptr, tC, 6, D, D, 0);
  gemm(path, af_wq, nullptr, nullptr, tA, 2 * N, D, D, 0);
  flash_attn_kernel<<<dim3((2 * N) / 16, 8), 32, 0, stream>>>(tA, tB, tC, aO, 2 * N, 6);
  gemm(aO, af_wo, nullptr, path, tok + 6 * D, 2 * N, D, D, 0);

  // ---- FFN residual + LayerNorm ----
  const int TOKN = 2 * N + 6;
  gemm(tok, ffn_w1, ffn_b1, nullptr, hid, TOKN, D, 1024, 1);
  gemm(hid, ffn_w2, ffn_b2, tok, tok, TOKN, 1024, D, 0);
  layernorm_kernel<<<TOKN, 256, 0, stream>>>(tok, ln_g, ln_b);

  // ---- embeds + classifier head ----
  mean_embed_kernel<<<1, 256, 0, stream>>>(tok, ge, pe);
  head_kernel<<<1, 128, 0, stream>>>(ge, pe, mm_w, mm_b, cls_w, cls_b,
                                     (float*)d_out);
}
